// Attention_28406913696361
// MI455X (gfx1250) — matrix operand order repaired and checked by reading, not compiled, not run on hardware
//
#include <hip/hip_runtime.h>

typedef __attribute__((ext_vector_type(16))) _Float16 v16h;
typedef __attribute__((ext_vector_type(8)))  float    v8f;
typedef __attribute__((ext_vector_type(4)))  unsigned uint4v;
typedef __attribute__((ext_vector_type(4)))  float    float4v;

__device__ __forceinline__ v8f wmma_f16(v16h a, v16h b, v8f c) {
  return __builtin_amdgcn_wmma_f32_16x16x32_f16(false, a, false, b, (short)0, c,
                                                false, false);
}

// Async global->LDS copy (16B per lane), tracked with ASYNCcnt.
// LDS operand: low 32 bits of the generic shared-memory pointer (flat->LDS
// addressing truncates to addr[31:0] per ISA 10.2).
__device__ __forceinline__ void async_b128(void* lds, const void* g) {
  unsigned l = (unsigned)(unsigned long long)lds;
  unsigned long long a = (unsigned long long)g;
  asm volatile("global_load_async_to_lds_b128 %0, %1, off"
               :: "v"(l), "v"(a) : "memory");
}
template <int N>
__device__ __forceinline__ void wait_asynccnt() {
  asm volatile("s_wait_asynccnt %0" :: "n"(N) : "memory");
}

// A-fragment (16x32 f16) from LDS tile stored [m][k], stride in halfs.
// Lane L: m = L%16, hi = L/16; half t holds K = (t/8)*16 + hi*8 + (t%8).
__device__ __forceinline__ v16h ld_fragA(const _Float16* tile, int row0,
                                         int stride, int lane) {
  const _Float16* p = tile + (row0 + (lane & 15)) * stride + ((lane >> 4) << 3);
  union { v16h v; unsigned u[8]; } f;
#pragma unroll
  for (int g = 0; g < 2; ++g)
#pragma unroll
    for (int j = 0; j < 4; ++j)
      f.u[g * 4 + j] = *(const unsigned*)(p + g * 16 + 2 * j);
  return f.v;
}

// B-fragment (32x16 f16) from LDS tile stored TRANSPOSED as [n][k].
// Lane L: n = L%16; half t holds K = (t/8)*16 + (L/16)*8 + (t%8).
__device__ __forceinline__ v16h ld_fragB(const _Float16* tile, int row0,
                                         int stride, int lane) {
  const _Float16* p = tile + (row0 + (lane & 15)) * stride + ((lane >> 4) << 3);
  union { v16h v; unsigned u[8]; } f;
#pragma unroll
  for (int j = 0; j < 8; ++j)
    f.u[j] = *(const unsigned*)(p + (j >> 2) * 16 + 2 * (j & 3));
  return f.v;
}

// ---------------- weight prep kernels ----------------
__global__ void k_cvt(const float* __restrict__ s, _Float16* __restrict__ d, int n) {
  int i = blockIdx.x * blockDim.x + threadIdx.x;
  if (i < n) d[i] = (_Float16)s[i];
}
// dst[c*R + r] = src[r*C + c]   (make Bt = [N,K] from [K,N])
__global__ void k_cvtT(const float* __restrict__ s, _Float16* __restrict__ d,
                       int R, int C) {
  int i = blockIdx.x * blockDim.x + threadIdx.x;
  if (i < R * C) { int r = i / C, c = i % C; d[c * R + r] = (_Float16)s[i]; }
}
// Wo[h,dcol,e] -> Bt[n=dcol][k=h*64+e]
__global__ void k_cvt_wo(const float* __restrict__ wo, _Float16* __restrict__ d) {
  int i = blockIdx.x * blockDim.x + threadIdx.x;
  if (i < 512 * 512) {
    int n = i >> 9, k = i & 511;
    d[i] = (_Float16)wo[(k >> 6) * (512 * 64) + n * 64 + (k & 63)];
  }
}

// ---------------- generic WMMA GEMM (async double-buffered) ----------------
// C[M,N] = op(A[M,K] @ Bt[N,K]^T + bias), A fp32-or-f16, out fp32-or-f16.
constexpr int BM = 128, BN = 128, BK = 32, ASTR = 48;

template <bool AF32, bool OF32>
__global__ void __launch_bounds__(256)
gemm_wmma(const void* __restrict__ Aptr, const _Float16* __restrict__ Bt,
          const float* __restrict__ bias, void* __restrict__ Outp,
          int M, int N, int K, int do_relu) {
  __shared__ _Float16 As[2][BM * ASTR];
  __shared__ _Float16 Bs[2][BN * ASTR];
  const int tid = threadIdx.x;
  const int lane = tid & 31;
  const int wid = tid >> 5;
  const int wm = wid >> 2, wn = wid & 3;           // 2 x 4 wave grid
  const int m0 = blockIdx.y * BM, n0 = blockIdx.x * BN;
  const int nt = K / BK;
  // async instructions issued per wave per stage (in-order completion)
  constexpr int PS = AF32 ? 2 : 4;

  auto stage = [&](int kt, int buf) {
    const int k0 = kt * BK;
    if (AF32) {
      const float* Ag = (const float*)Aptr;
#pragma unroll
      for (int c = 0; c < 4; ++c) {
        int flat = tid * 16 + c * 4;
        int row = flat >> 5, col = flat & 31;
        float4v v = *(const float4v*)(Ag + (size_t)(m0 + row) * K + k0 + col);
        union { _Float16 h[4]; unsigned u[2]; } t;
        t.h[0] = (_Float16)v[0]; t.h[1] = (_Float16)v[1];
        t.h[2] = (_Float16)v[2]; t.h[3] = (_Float16)v[3];
        unsigned* dst = (unsigned*)&As[buf][row * ASTR + col];
        dst[0] = t.u[0]; dst[1] = t.u[1];
      }
    } else {
      const _Float16* Ag = (const _Float16*)Aptr;
#pragma unroll
      for (int c = 0; c < 2; ++c) {
        int flat = tid * 16 + c * 8;
        int row = flat >> 5, col = flat & 31;
        async_b128(&As[buf][row * ASTR + col],
                   Ag + (size_t)(m0 + row) * K + k0 + col);
      }
    }
#pragma unroll
    for (int c = 0; c < 2; ++c) {
      int flat = tid * 16 + c * 8;
      int row = flat >> 5, col = flat & 31;
      async_b128(&Bs[buf][row * ASTR + col],
                 Bt + (size_t)(n0 + row) * K + k0 + col);
    }
  };

  v8f zero = {};
  v8f acc[4][2];
#pragma unroll
  for (int i = 0; i < 4; ++i)
#pragma unroll
    for (int j = 0; j < 2; ++j) acc[i][j] = zero;

  stage(0, 0);
  for (int kt = 0; kt < nt; ++kt) {
    const int cur = kt & 1;
    const bool has_next = (kt + 1) < nt;
    if (has_next) {
      stage(kt + 1, cur ^ 1);     // prefetch next tile into other buffer
      wait_asynccnt<PS>();        // current tile's loads complete (in order)
    } else {
      wait_asynccnt<0>();
    }
    __syncthreads();

    v16h bfr[2];
#pragma unroll
    for (int nf = 0; nf < 2; ++nf)
      bfr[nf] = ld_fragB(Bs[cur], wn * 32 + nf * 16, ASTR, lane);
#pragma unroll
    for (int mf = 0; mf < 4; ++mf) {
      v16h afr = ld_fragA(As[cur], wm * 64 + mf * 16, ASTR, lane);
#pragma unroll
      for (int nf = 0; nf < 2; ++nf)
        acc[mf][nf] = wmma_f16(afr, bfr[nf], acc[mf][nf]);
    }
    __syncthreads();
  }

#pragma unroll
  for (int mf = 0; mf < 4; ++mf)
#pragma unroll
    for (int nf = 0; nf < 2; ++nf) {
      int col = n0 + wn * 32 + nf * 16 + (lane & 15);
      float bv = bias ? bias[col] : 0.0f;
      int rbase = m0 + wm * 64 + mf * 16 + ((lane >> 4) << 3);
#pragma unroll
      for (int j = 0; j < 8; ++j) {
        float v = acc[mf][nf][j] + bv;
        if (do_relu) v = fmaxf(v, 0.0f);
        if (OF32)
          ((float*)Outp)[(size_t)(rbase + j) * N + col] = v;
        else
          ((_Float16*)Outp)[(size_t)(rbase + j) * N + col] = (_Float16)v;
      }
    }
}

// ---------------- flash attention ----------------
// grid (n/64, H, B), 128 threads. Q/K/V/O are [B*n, 512] f16, head h at cols h*64..
__global__ void __launch_bounds__(128)
flash_attn(const _Float16* __restrict__ Q, const _Float16* __restrict__ Km,
           const _Float16* __restrict__ V, _Float16* __restrict__ O) {
  constexpr int D = 512, HD = 64, SEQ = 2048;
  constexpr int QS = 72;  // f16 LDS stride (halfs)
  constexpr int SS = 68;  // f32 LDS stride
  __shared__ _Float16 Qs[64 * QS];
  __shared__ _Float16 Ks[64 * QS];
  __shared__ _Float16 Vt[64 * QS];  // [e][nk] (transposed V block)
  __shared__ _Float16 Ps[64 * QS];
  __shared__ float    Sc[64 * SS];
  __shared__ float mrow[64], lrow[64], rfac[64];

  const int tid = threadIdx.x;
  const int lane = tid & 31;
  const int w = tid >> 5;  // 0..3, wave handles query rows [w*16, w*16+16)
  const int qb = blockIdx.x, h = blockIdx.y, b = blockIdx.z;

  const _Float16* Qg = Q + ((size_t)b * SEQ + qb * 64) * D + h * HD;
  const _Float16* Kg = Km + (size_t)b * SEQ * D + h * HD;
  const _Float16* Vg = V + (size_t)b * SEQ * D + h * HD;

  {  // async load 64x64 Q tile (completion covered by first in-loop wait)
    int row = tid >> 1, cb = (tid & 1) * 32;
#pragma unroll
    for (int c = 0; c < 4; ++c)
      async_b128(&Qs[row * QS + cb + c * 8], Qg + (size_t)row * D + cb + c * 8);
  }
  if (tid < 64) { mrow[tid] = -3.0e38f; lrow[tid] = 0.0f; }

  v8f zero = {};
  v8f acc[4];
#pragma unroll
  for (int i = 0; i < 4; ++i) acc[i] = zero;

  for (int kb = 0; kb < SEQ / 64; ++kb) {
    {  // async stage K block [nk][e]
      int row = tid >> 1, cb = (tid & 1) * 32;
#pragma unroll
      for (int c = 0; c < 4; ++c)
        async_b128(&Ks[row * QS + cb + c * 8],
                   Kg + (size_t)(kb * 64 + row) * D + cb + c * 8);
    }
    {  // stage V block transposed -> Vt[e][nk] (scatter, sync path)
      int nk = tid >> 1, eb = (tid & 1) * 32;
      union { uint4v q[4]; _Float16 hh[32]; } buf;
#pragma unroll
      for (int c = 0; c < 4; ++c)
        buf.q[c] = *(const uint4v*)(Vg + (size_t)(kb * 64 + nk) * D + eb + c * 8);
#pragma unroll
      for (int i = 0; i < 32; ++i) Vt[(eb + i) * QS + nk] = buf.hh[i];
    }
    wait_asynccnt<0>();
    __syncthreads();

    // S = (Q K^T) * 1/sqrt(64)
    v16h qfr[2];
#pragma unroll
    for (int kc = 0; kc < 2; ++kc)
      qfr[kc] = ld_fragA(Qs + kc * 32, w * 16, QS, lane);
#pragma unroll
    for (int nf = 0; nf < 4; ++nf) {
      v8f s = zero;
#pragma unroll
      for (int kc = 0; kc < 2; ++kc) {
        v16h kfr = ld_fragB(Ks + kc * 32, nf * 16, QS, lane);
        s = wmma_f16(qfr[kc], kfr, s);
      }
      int col = nf * 16 + (lane & 15);
      int r0 = w * 16 + ((lane >> 4) << 3);
#pragma unroll
      for (int j = 0; j < 8; ++j) Sc[(r0 + j) * SS + col] = s[j] * 0.125f;
    }
    __syncthreads();

    // online softmax per query row
    if (tid < 64) {
      int rr = tid;
      float mo = mrow[rr], mx = mo;
      for (int j = 0; j < 64; ++j) mx = fmaxf(mx, Sc[rr * SS + j]);
      float rf = __expf(mo - mx);
      float sum = 0.0f;
      for (int j = 0; j < 64; ++j) {
        float p = __expf(Sc[rr * SS + j] - mx);
        sum += p;
        Ps[rr * QS + j] = (_Float16)p;
      }
      mrow[rr] = mx;
      lrow[rr] = lrow[rr] * rf + sum;
      rfac[rr] = rf;
    }
    __syncthreads();

    // rescale register accumulators, then O += P @ V
    float rl[8];
    {
      int r0 = w * 16 + ((lane >> 4) << 3);
#pragma unroll
      for (int j = 0; j < 8; ++j) rl[j] = rfac[r0 + j];
    }
#pragma unroll
    for (int nf = 0; nf < 4; ++nf)
#pragma unroll
      for (int j = 0; j < 8; ++j) acc[nf][j] = acc[nf][j] * rl[j];

    v16h pfr[2];
#pragma unroll
    for (int kc = 0; kc < 2; ++kc)
      pfr[kc] = ld_fragA(Ps + kc * 32, w * 16, QS, lane);
#pragma unroll
    for (int nf = 0; nf < 4; ++nf)
#pragma unroll
      for (int kc = 0; kc < 2; ++kc) {
        v16h vfr = ld_fragB(Vt + kc * 32, nf * 16, QS, lane);
        acc[nf] = wmma_f16(pfr[kc], vfr, acc[nf]);
      }
    __syncthreads();
  }

  if (tid < 64) rfac[tid] = 1.0f / lrow[tid];
  __syncthreads();

  _Float16* Og = O + ((size_t)b * SEQ + qb * 64) * D + h * HD;
  int r0 = w * 16 + ((lane >> 4) << 3);
  int c0 = lane & 15;
#pragma unroll
  for (int nf = 0; nf < 4; ++nf)
#pragma unroll
    for (int j = 0; j < 8; ++j)
      Og[(size_t)(r0 + j) * D + nf * 16 + c0] = (_Float16)(acc[nf][j] * rfac[r0 + j]);
}

// ---------------- launcher ----------------
extern "C" void kernel_launch(void* const* d_in, const int* in_sizes, int n_in,
                              void* d_out, int out_size, void* d_ws, size_t ws_size,
                              hipStream_t stream) {
  (void)in_sizes; (void)n_in; (void)out_size;
  const float* x1 = (const float*)d_in[0];
  const float* x2 = (const float*)d_in[1];
  const float* r  = (const float*)d_in[2];
  const float* W1 = (const float*)d_in[3];
  const float* b1 = (const float*)d_in[4];
  const float* W2 = (const float*)d_in[5];
  const float* b2 = (const float*)d_in[6];
  const float* Wq = (const float*)d_in[7];
  const float* Wk = (const float*)d_in[8];
  const float* Wv = (const float*)d_in[9];
  const float* Wo = (const float*)d_in[10];
  float* out = (float*)d_out;

  const int Bt = 8, n = 2048, dmod = 512, dx = 128;
  const int M = Bt * n;  // 16384

  char* ws = (char*)d_ws;
  size_t off = 0;
  auto alloc = [&](size_t bytes) {
    off = (off + 255) & ~(size_t)255;
    char* p = ws + off;
    off += bytes;
    return p;
  };
  _Float16* W1t = (_Float16*)alloc((size_t)dmod * dx * 2);    // [512,128]
  _Float16* W2t = (_Float16*)alloc((size_t)dmod * dmod * 2);  // [512,512]
  _Float16* Wqh = (_Float16*)alloc((size_t)dmod * dmod * 2);
  _Float16* Wkh = (_Float16*)alloc((size_t)dmod * dmod * 2);
  _Float16* Wvh = (_Float16*)alloc((size_t)dmod * dmod * 2);
  _Float16* Wot = (_Float16*)alloc((size_t)dmod * dmod * 2);
  _Float16* bufE = (_Float16*)alloc((size_t)M * dmod * 2);  // h1 -> qh
  _Float16* bufF = (_Float16*)alloc((size_t)M * dmod * 2);  // h2 -> kh
  _Float16* bufG = (_Float16*)alloc((size_t)M * dmod * 2);  // kM -> O
  _Float16* bufH = (_Float16*)alloc((size_t)M * dmod * 2);  // qM
  _Float16* bufV = (_Float16*)alloc((size_t)M * dmod * 2);  // vh
  if (off > ws_size) return;

  // weight prep
  k_cvtT<<<(dx * dmod + 255) / 256, 256, 0, stream>>>(W1, W1t, dx, dmod);
  k_cvtT<<<(dmod * dmod + 255) / 256, 256, 0, stream>>>(W2, W2t, dmod, dmod);
  k_cvt<<<(dmod * dmod + 255) / 256, 256, 0, stream>>>(Wq, Wqh, dmod * dmod);
  k_cvt<<<(dmod * dmod + 255) / 256, 256, 0, stream>>>(Wk, Wkh, dmod * dmod);
  k_cvt<<<(dmod * dmod + 255) / 256, 256, 0, stream>>>(Wv, Wvh, dmod * dmod);
  k_cvt_wo<<<(dmod * dmod + 255) / 256, 256, 0, stream>>>(Wo, Wot);

  dim3 g(dmod / BN, M / BM), blk(256);
  // MLP layer 1 (fp32 A, relu)
  gemm_wmma<true, false><<<g, blk, 0, stream>>>(x1, W1t, b1, bufE, M, dmod, dx, 1);
  gemm_wmma<true, false><<<g, blk, 0, stream>>>(x2, W1t, b1, bufF, M, dmod, dx, 1);
  // MLP layer 2
  gemm_wmma<false, false><<<g, blk, 0, stream>>>(bufE, W2t, b2, bufG, M, dmod, dmod, 0);  // kM
  gemm_wmma<false, false><<<g, blk, 0, stream>>>(bufF, W2t, b2, bufH, M, dmod, dmod, 0);  // qM
  // head projections (all heads fused into one 512x512 GEMM each)
  gemm_wmma<false, false><<<g, blk, 0, stream>>>(bufH, Wqh, nullptr, bufE, M, dmod, dmod, 0);  // qh
  gemm_wmma<false, false><<<g, blk, 0, stream>>>(bufG, Wkh, nullptr, bufF, M, dmod, dmod, 0);  // kh
  gemm_wmma<true,  false><<<g, blk, 0, stream>>>(r,    Wvh, nullptr, bufV, M, dmod, dmod, 0);  // vh
  // flash attention, O into bufG
  flash_attn<<<dim3(n / 64, 8, Bt), 128, 0, stream>>>(bufE, bufF, bufV, bufG);
  // output projection (sum over heads), fp32 out
  gemm_wmma<false, true><<<g, blk, 0, stream>>>(bufG, Wot, nullptr, out, M, dmod, dmod, 0);
}
